// AttnDecoderRNN_40656160424712
// MI455X (gfx1250) — matrix operand order repaired and checked by reading, not compiled, not run on hardware
//
#include <hip/hip_runtime.h>
#include <hip/hip_bf16.h>
#include <math.h>

#define HID   1024
#define BATCH 1024
#define SRC   10
#define VOCAB 50257

typedef __bf16 bf16_t;
typedef __attribute__((ext_vector_type(16))) __bf16 v16bf;
typedef __attribute__((ext_vector_type(8)))  __bf16 v8bf;
typedef __attribute__((ext_vector_type(8)))  float  v8f;
typedef __attribute__((ext_vector_type(4)))  float  v4f;

#define WAVES 8   // waves per block in GEMM (one N-tile each)
#define MT    32  // 16-row M-tiles per wave (512 rows persistent accumulation)

// ---------------------------------------------------------------------------
// prep: embedding gather + f32 -> bf16 for x and h0
// ---------------------------------------------------------------------------
__global__ void prep_kernel(const int* __restrict__ idx,
                            const float* __restrict__ emb,
                            const float* __restrict__ h0,
                            bf16_t* __restrict__ x_bf,
                            bf16_t* __restrict__ h0_bf) {
  int t = blockIdx.x * blockDim.x + threadIdx.x;   // over BATCH * HID/4
  int b = t >> 8;             // HID/4 == 256
  int k = (t & 255) * 4;
  int row = idx[b];
  v4f xe = *(const v4f*)(emb + (size_t)row * HID + k);
  v4f hv = *(const v4f*)(h0  + (size_t)b   * HID + k);
#pragma unroll
  for (int j = 0; j < 4; ++j) {
    x_bf [(size_t)b * HID + k + j] = (bf16_t)xe[j];
    h0_bf[(size_t)b * HID + k + j] = (bf16_t)hv[j];
  }
}

// ---------------------------------------------------------------------------
// Generic WMMA GEMM: C[m][n] = sum_k A_bf16[m][k] * W_f32[n][k] (+ bias[n])
// A: M x K bf16 row-major (K = HID), W: N x K f32 row-major.
// grid.x covers N-tiles (WAVES tiles per block), grid.y covers M in 512-row
// halves. Each wave: 1 N-tile x 32 M-tiles of persistent f32 accumulators.
// ---------------------------------------------------------------------------
__global__ __launch_bounds__(256)
void wmma_gemm_bf16(const bf16_t* __restrict__ A,
                    const float*  __restrict__ W,
                    const float*  __restrict__ bias,
                    float* __restrict__ C,
                    int N, int ldc) {
  const int K = HID;
  const int lane = threadIdx.x & 31;
  const int wave = threadIdx.x >> 5;
  const int l    = lane & 15;
  const int half = lane >> 4;

  const int nTile = blockIdx.x * WAVES + wave;
  const int n0    = nTile * 16;

  // B-matrix (32x16, K-major in lanes): lane l<16 -> column n0+l with K 0..15,
  // lane l>=16 -> column n0+l-16 with K 16..31. Column n of B == row n of W.
  int wr = n0 + l;
  if (wr > N - 1) wr = N - 1;                  // clamp OOB rows (stores masked)
  const float* wrow = W + (size_t)wr * K + (half ? 16 : 0);

  // A-matrix (16x32): lanes 0..15 hold row m0+l with K 0..15,
  // lanes 16..31 hold the same rows with K 16..31.
  const int m0 = blockIdx.y * (MT * 16);
  const bf16_t* abase = A + (size_t)(m0 + l) * K + half * 16;

  v8f acc[MT] = {};

#pragma unroll 1
  for (int kb = 0; kb < K; kb += 32) {
    const float* wp = wrow + kb;
    __builtin_prefetch(wp + 32, 0, 1);         // global_prefetch_b8, next chunk
    v4f w0 = *(const v4f*)(wp + 0);
    v4f w1 = *(const v4f*)(wp + 4);
    v4f w2 = *(const v4f*)(wp + 8);
    v4f w3 = *(const v4f*)(wp + 12);
    v16bf bfrag;
#pragma unroll
    for (int j = 0; j < 4; ++j) {
      bfrag[j]      = (bf16_t)w0[j];
      bfrag[4 + j]  = (bf16_t)w1[j];
      bfrag[8 + j]  = (bf16_t)w2[j];
      bfrag[12 + j] = (bf16_t)w3[j];
    }

    const bf16_t* ap = abase + kb;
#pragma unroll
    for (int mt = 0; mt < MT; ++mt) {
      v8bf alo = *(const v8bf*)(ap);
      v8bf ahi = *(const v8bf*)(ap + 8);
      v16bf afrag;
#pragma unroll
      for (int j = 0; j < 8; ++j) { afrag[j] = alo[j]; afrag[8 + j] = ahi[j]; }
      acc[mt] = __builtin_amdgcn_wmma_f32_16x16x32_bf16(
          false, afrag, false, bfrag, (short)0, acc[mt], false, false);
      ap += (size_t)16 * K;
    }
  }

  // D layout: VGPR r -> M = r + half*8, N = n0 + (lane & 15)
  const int ncol = n0 + l;
  const bool nok = ncol < N;
  const float bv = (nok && bias) ? bias[ncol] : 0.0f;
#pragma unroll
  for (int mt = 0; mt < MT; ++mt) {
    const int rbase = m0 + mt * 16 + half * 8;
#pragma unroll
    for (int r = 0; r < 8; ++r) {
      if (nok) C[(size_t)(rbase + r) * ldc + ncol] = acc[mt][r] + bv;
    }
  }
}

// ---------------------------------------------------------------------------
// GRU gate combine: h = (1-z)*n + z*h0  (PyTorch gate order r,z,n)
// ---------------------------------------------------------------------------
__global__ void gru_combine(const float* __restrict__ gi,
                            const float* __restrict__ gh,
                            const float* __restrict__ b_ih,
                            const float* __restrict__ b_hh,
                            const float* __restrict__ h0,
                            float* __restrict__ h_out,
                            bf16_t* __restrict__ h_bf) {
  int t = blockIdx.x * blockDim.x + threadIdx.x;  // B*H
  int b = t >> 10;
  int i = t & 1023;
  size_t base = (size_t)b * 3 * HID;
  float gr = gi[base + i]           + b_ih[i]           + gh[base + i]           + b_hh[i];
  float gz = gi[base + HID + i]     + b_ih[HID + i]     + gh[base + HID + i]     + b_hh[HID + i];
  float r = 1.0f / (1.0f + expf(-gr));
  float z = 1.0f / (1.0f + expf(-gz));
  float n = tanhf(gi[base + 2*HID + i] + b_ih[2*HID + i] +
                  r * (gh[base + 2*HID + i] + b_hh[2*HID + i]));
  float hv = (1.0f - z) * n + z * h0[(size_t)b * HID + i];
  h_out[(size_t)b * HID + i] = hv;
  h_bf [(size_t)b * HID + i] = (bf16_t)hv;
}

// ---------------------------------------------------------------------------
// Attention energies + masked softmax over S=10. One block (10 waves) per b.
// ---------------------------------------------------------------------------
__global__ void attn_kernel(const float* __restrict__ h,
                            const float* __restrict__ enc,
                            const unsigned char* __restrict__ mask,
                            float* __restrict__ attn) {
  int b = blockIdx.x;
  int wave = threadIdx.x >> 5;
  int lane = threadIdx.x & 31;
  __shared__ float e_sh[SRC];
  if (wave < SRC) {
    const float* hp = h + (size_t)b * HID;
    const float* ep = enc + ((size_t)b * SRC + wave) * HID;
    float acc = 0.0f;
    for (int k = lane * 4; k < HID; k += 32 * 4) {
      v4f hv = *(const v4f*)(hp + k);
      v4f ev = *(const v4f*)(ep + k);
      acc += hv[0]*ev[0] + hv[1]*ev[1] + hv[2]*ev[2] + hv[3]*ev[3];
    }
    for (int off = 16; off > 0; off >>= 1) acc += __shfl_down(acc, off);
    if (lane == 0) e_sh[wave] = acc;
  }
  __syncthreads();
  if (threadIdx.x == 0) {
    float e[SRC];
    float mx = -INFINITY;
    for (int s = 0; s < SRC; ++s) {
      float v = mask[(size_t)b * SRC + s] ? e_sh[s] : -1e9f;
      e[s] = v;
      mx = fmaxf(mx, v);
    }
    float sum = 0.0f;
    for (int s = 0; s < SRC; ++s) { e[s] = expf(e[s] - mx); sum += e[s]; }
    float inv = 1.0f / sum;
    for (int s = 0; s < SRC; ++s) attn[(size_t)b * SRC + s] = e[s] * inv;
  }
}

// ---------------------------------------------------------------------------
// Per-row logsumexp over VOCAB. One block per row.
// ---------------------------------------------------------------------------
__global__ void lse_kernel(const float* __restrict__ logits,
                           float* __restrict__ lse) {
  int b = blockIdx.x;
  const float* row = logits + (size_t)b * VOCAB;
  __shared__ float red[256];
  float mx = -INFINITY;
  for (int v = threadIdx.x; v < VOCAB; v += 256) mx = fmaxf(mx, row[v]);
  red[threadIdx.x] = mx;
  __syncthreads();
  for (int s = 128; s > 0; s >>= 1) {
    if (threadIdx.x < s) red[threadIdx.x] = fmaxf(red[threadIdx.x], red[threadIdx.x + s]);
    __syncthreads();
  }
  mx = red[0];
  __syncthreads();
  float sum = 0.0f;
  for (int v = threadIdx.x; v < VOCAB; v += 256) sum += expf(row[v] - mx);
  red[threadIdx.x] = sum;
  __syncthreads();
  for (int s = 128; s > 0; s >>= 1) {
    if (threadIdx.x < s) red[threadIdx.x] += red[threadIdx.x + s];
    __syncthreads();
  }
  if (threadIdx.x == 0) lse[b] = mx + logf(red[0]);
}

// ---------------------------------------------------------------------------
// output = logits - lse[b], in place
// ---------------------------------------------------------------------------
__global__ void finalize_kernel(float* __restrict__ out,
                                const float* __restrict__ lse) {
  size_t i = (size_t)blockIdx.x * blockDim.x + threadIdx.x;
  if (i < (size_t)BATCH * VOCAB) {
    int b = (int)(i / VOCAB);
    out[i] -= lse[b];
  }
}

// ---------------------------------------------------------------------------
extern "C" void kernel_launch(void* const* d_in, const int* in_sizes, int n_in,
                              void* d_out, int out_size, void* d_ws, size_t ws_size,
                              hipStream_t stream) {
  const int*           input_step  = (const int*)  d_in[0];
  const float*         last_hidden = (const float*)d_in[1];
  const float*         enc         = (const float*)d_in[2];
  const unsigned char* mask        = (const unsigned char*)d_in[3];
  const float*         emb         = (const float*)d_in[4];
  const float*         w_ih        = (const float*)d_in[5];
  const float*         w_hh        = (const float*)d_in[6];
  const float*         b_ih        = (const float*)d_in[7];
  const float*         b_hh        = (const float*)d_in[8];
  const float*         out_w       = (const float*)d_in[9];
  const float*         out_b       = (const float*)d_in[10];

  char* ws = (char*)d_ws;
  bf16_t* x_bf  = (bf16_t*)(ws + (size_t)0);
  bf16_t* h0_bf = (bf16_t*)(ws + ((size_t)2  << 20));
  bf16_t* h_bf  = (bf16_t*)(ws + ((size_t)4  << 20));
  float*  gi    = (float*) (ws + ((size_t)6  << 20));   // B x 3H
  float*  gh    = (float*) (ws + ((size_t)18 << 20));   // B x 3H
  float*  lse   = (float*) (ws + ((size_t)30 << 20));   // B

  float* out_logits = (float*)d_out;                       // B x V
  float* out_hidden = out_logits + (size_t)BATCH * VOCAB;  // 1 x B x H
  float* out_attn   = out_hidden + (size_t)BATCH * HID;    // B x S

  // 1) embedding gather + bf16 conversion of x, h0
  prep_kernel<<<BATCH, 256, 0, stream>>>(input_step, emb, last_hidden, x_bf, h0_bf);

  // 2,3) GRU input/hidden GEMMs: (1024 x 3072 x 1024) each, WMMA bf16
  {
    int tilesN = (3 * HID) / 16;                 // 192
    dim3 grid((tilesN + WAVES - 1) / WAVES, BATCH / (MT * 16));
    wmma_gemm_bf16<<<grid, 256, 0, stream>>>(x_bf,  w_ih, nullptr, gi, 3 * HID, 3 * HID);
    wmma_gemm_bf16<<<grid, 256, 0, stream>>>(h0_bf, w_hh, nullptr, gh, 3 * HID, 3 * HID);
  }

  // 4) GRU gate fusion -> h (f32 to output hidden slot, bf16 to workspace)
  gru_combine<<<(BATCH * HID) / 256, 256, 0, stream>>>(gi, gh, b_ih, b_hh,
                                                       last_hidden, out_hidden, h_bf);

  // 5) attention energies + softmax over S
  attn_kernel<<<BATCH, 320, 0, stream>>>(out_hidden, enc, mask, out_attn);

  // 6) output projection: (1024 x 50257 x 1024) WMMA bf16, bias fused
  {
    int tilesN = (VOCAB + 15) / 16;              // 3142
    dim3 grid((tilesN + WAVES - 1) / WAVES, BATCH / (MT * 16));
    wmma_gemm_bf16<<<grid, 256, 0, stream>>>(h_bf, out_w, out_b, out_logits,
                                             VOCAB, VOCAB);
  }

  // 7) per-row logsumexp, 8) subtract
  lse_kernel<<<BATCH, 256, 0, stream>>>(out_logits, lse);
  size_t total = (size_t)BATCH * VOCAB;
  finalize_kernel<<<(unsigned)((total + 255) / 256), 256, 0, stream>>>(out_logits, lse);
}
